// RelativeSelfAttention_89489938579603
// MI455X (gfx1250) — matrix operand order repaired and checked by reading, not compiled, not run on hardware
//
#include <hip/hip_runtime.h>
#include <hip/hip_bf16.h>
#include <math.h>
#include <stdint.h>

// ---------------------------------------------------------------------------
// Relative self-attention forward for MI455X (gfx1250), wave32 + WMMA bf16.
//   B=2, T=2048, C=1024, H=16, D=64, MAXREL=32
// Outputs: y [B,T,C] fp32  ++  post_soft_att [B,H,T,T] fp32 (concat in d_out)
// ---------------------------------------------------------------------------

#define B_    2
#define T_    2048
#define C_    1024
#define H_    16
#define D_    64
#define NREL  65          // 2*MAXREL+1
#define C3_   (3 * C_)
#define PES   (T_ + 8)    // padded bf16 strip stride (bank spread)

typedef __bf16 bf16;
typedef __attribute__((ext_vector_type(16))) __bf16 v16bf;
typedef __attribute__((ext_vector_type(8)))  float  v8f;
typedef __attribute__((ext_vector_type(4)))  float  f32x4;
typedef __attribute__((ext_vector_type(4)))  unsigned int u32x4;
typedef __attribute__((ext_vector_type(8)))  int    i32x8;
typedef __attribute__((ext_vector_type(4)))  int    i32x4;

// D = A(16x32 bf16) * B(32x16 bf16) + C(16x16 f32)
__device__ __forceinline__ v8f wmma_bf16(v16bf a, v16bf b, v8f c) {
  return __builtin_amdgcn_wmma_f32_16x16x32_bf16(
      false, a, false, b, (short)0, c, false, false);
}

// A fragment 16x32 (MxK) from row-major src, leading dim ld (contiguous 2x16B).
__device__ __forceinline__ v16bf load_a_row_major(const bf16* __restrict__ p0,
                                                  int ld, int lane) {
  const int row = lane & 15, half = lane >> 4;
  const bf16* p = p0 + (size_t)row * ld + half * 16;
  v16bf a;
#pragma unroll
  for (int e = 0; e < 8; ++e) a[e] = p[e];
#pragma unroll
  for (int e = 0; e < 8; ++e) a[8 + e] = p[8 + e];
  return a;
}

// B fragment 32x16 (KxN), source stored N-major (B^T row-major, leading dim
// ld): per-lane data is one CONTIGUOUS 32B run.
__device__ __forceinline__ v16bf load_b_nmajor(const bf16* __restrict__ p0,
                                               int ld, int lane) {
  const int n = lane & 15, half = lane >> 4;
  const bf16* p = p0 + (size_t)n * ld + half * 16;
  v16bf b;
#pragma unroll
  for (int e = 0; e < 16; ++e) b[e] = p[e];
  return b;
}

__device__ __forceinline__ v16bf load_b_nmajor_lds(const bf16* p0, int ld,
                                                   int lane) {
  const int n = lane & 15, half = lane >> 4;
  const bf16* p = p0 + (size_t)n * ld + half * 16;
  v16bf b;
#pragma unroll
  for (int e = 0; e < 16; ++e) b[e] = p[e];
  return b;
}

// A fragment from the LDS bf16 exp strip (row stride PES).
__device__ __forceinline__ v16bf load_a_pe(const bf16* pe_base, int k0,
                                           int lane) {
  const int rowm = lane & 15, half = lane >> 4;
  const bf16* p = pe_base + rowm * PES + k0 + half * 16;
  v16bf a;
#pragma unroll
  for (int e = 0; e < 8; ++e) a[e] = p[e];
#pragma unroll
  for (int e = 0; e < 8; ++e) a[8 + e] = p[8 + e];
  return a;
}

// --------------------- CDNA5 async global->LDS (ASYNCcnt) ------------------
__device__ __forceinline__ void async_load_b128(uint32_t lds_addr,
                                                const void* gaddr) {
  asm volatile("global_load_async_to_lds_b128 %0, %1, off"
               :: "v"(lds_addr), "v"(gaddr) : "memory");
}
__device__ __forceinline__ void wait_async0() {
  asm volatile("s_wait_asynccnt 0" ::: "memory");
}
__device__ __forceinline__ void wait_async4() {
  asm volatile("s_wait_asynccnt 4" ::: "memory");
}

// ------------------ CDNA5 Tensor Data Mover (TENSORcnt) --------------------
// 1-D contiguous tile load, D# built per CDNA5 ISA #8.
// 6-arg builtin form: (uint32x4 g0, int32x8 g1, int32x4, int32x4, int32x8, i32)
__device__ __forceinline__ void tdm_load_1d_dwords(uint32_t lds_addr,
                                                   const void* gaddr,
                                                   uint32_t ndwords) {
  const uint64_t ga = (uint64_t)(uintptr_t)gaddr;
  u32x4 g0;
  g0[0] = 1u;                                            // count=1
  g0[1] = lds_addr;
  g0[2] = (uint32_t)ga;
  g0[3] = ((uint32_t)(ga >> 32) & 0x01FFFFFFu) | 0x80000000u;  // type=2
  i32x8 g1;
  g1[0] = (int)(2u << 16);                               // data_size = 4 bytes
  g1[1] = (int)((ndwords & 0xFFFFu) << 16);              // tensor_dim0 lo16
  g1[2] = (int)(ndwords >> 16);                          // tensor_dim0 hi16
  g1[3] = (int)((ndwords & 0xFFFFu) << 16);              // tile_dim0
  g1[4] = 0;                                             // tile_dim1/2 = 0
  g1[5] = (int)ndwords;                                  // dim0 stride lo
  g1[6] = 0;
  g1[7] = 0;
  i32x4 z4 = {0, 0, 0, 0};
  i32x8 z8 = {0, 0, 0, 0, 0, 0, 0, 0};
  __builtin_amdgcn_tensor_load_to_lds(g0, g1, z4, z4, z8, 0);
}
__device__ __forceinline__ void wait_tensor0() {
  __builtin_amdgcn_s_wait_tensorcnt(0);
}

// ---------------------------------------------------------------------------
__global__ void f32_to_bf16_kernel(const float* __restrict__ in,
                                   bf16* __restrict__ out, int n) {
  int i = blockIdx.x * blockDim.x + threadIdx.x;
  int stride = gridDim.x * blockDim.x;
  for (; i < n; i += stride) out[i] = (bf16)in[i];
}

// wT[n*K + k] = (bf16) w[k*N + n]
__global__ void __launch_bounds__(256)
transpose_f32_to_bf16_kernel(const float* __restrict__ w, bf16* __restrict__ wT,
                             int K, int N) {
  __shared__ float tile[32][33];
  const int kb = blockIdx.y * 32, nb = blockIdx.x * 32;
  const int tx = threadIdx.x & 31, ty = threadIdx.x >> 5;
#pragma unroll
  for (int i = ty; i < 32; i += 8)
    tile[i][tx] = w[(size_t)(kb + i) * N + (nb + tx)];
  __syncthreads();
#pragma unroll
  for (int i = ty; i < 32; i += 8)
    wT[(size_t)(nb + i) * K + (kb + tx)] = (bf16)tile[tx][i];
}

// ---------------------------------------------------------------------------
// qkv = x @ w_attn + b_attn.  2x-unrolled software pipeline: two buffer sets
// alternate roles, so no register rotation (no v_mov walls / hazard NOPs).
__global__ void __launch_bounds__(128)
qkv_gemm_kernel(const bf16* __restrict__ xb,    // [B*T, C]
                const bf16* __restrict__ wT,    // [3C, C]
                const float* __restrict__ bias, // [3C]
                bf16* __restrict__ qh, bf16* __restrict__ kh,
                bf16* __restrict__ vt)          // vt: [B,H,D,T]
{
  const int lane = threadIdx.x & 31;
  const int wave = threadIdx.x >> 5;
  const int m0 = (blockIdx.y * 4 + wave) * 16;
  const int n0 = blockIdx.x * 64;
  v8f c[4] = {};
  const bf16* a_base = xb + (size_t)m0 * C_;
  const bf16* b_base[4];
#pragma unroll
  for (int j = 0; j < 4; ++j) b_base[j] = wT + (size_t)(n0 + j * 16) * C_;

  v16bf a0 = load_a_row_major(a_base, C_, lane);
  v16bf b0[4];
#pragma unroll
  for (int j = 0; j < 4; ++j) b0[j] = load_b_nmajor(b_base[j], C_, lane);
  v16bf a1, b1[4];

  for (int k0 = 0; k0 < C_; k0 += 64) {
    a1 = load_a_row_major(a_base + k0 + 32, C_, lane);
#pragma unroll
    for (int j = 0; j < 4; ++j)
      b1[j] = load_b_nmajor(b_base[j] + k0 + 32, C_, lane);
#pragma unroll
    for (int j = 0; j < 4; ++j) c[j] = wmma_bf16(a0, b0[j], c[j]);
    if (k0 + 64 < C_) {
      a0 = load_a_row_major(a_base + k0 + 64, C_, lane);
#pragma unroll
      for (int j = 0; j < 4; ++j)
        b0[j] = load_b_nmajor(b_base[j] + k0 + 64, C_, lane);
    }
#pragma unroll
    for (int j = 0; j < 4; ++j) c[j] = wmma_bf16(a1, b1[j], c[j]);
  }

#pragma unroll
  for (int j = 0; j < 4; ++j) {
    const int n = n0 + j * 16 + (lane & 15);
    const float bval = bias[n];
    const int which = n / C_;          // uniform per j-tile
    const int cc = n % C_;
    const int h = cc / D_, d = cc % D_;
#pragma unroll
    for (int r = 0; r < 8; ++r) {
      const int m = m0 + (lane >> 4) * 8 + r;
      const int bb = m >> 11, t = m & (T_ - 1);
      const float val = c[j][r] + bval;
      if (which == 0)
        qh[(((size_t)bb * H_ + h) * T_ + t) * D_ + d] = (bf16)val;
      else if (which == 1)
        kh[(((size_t)bb * H_ + h) * T_ + t) * D_ + d] = (bf16)val;
      else
        vt[(((size_t)bb * H_ + h) * D_ + d) * T_ + t] = (bf16)val;
    }
  }
}

// ---------------------------------------------------------------------------
__global__ void __launch_bounds__(128)
qrel_kernel(const bf16* __restrict__ qh, const float* __restrict__ relk,
            float* __restrict__ qrel) {
  __shared__ float qrow[D_];
  const size_t row = blockIdx.x;
  const int tid = threadIdx.x;
  if (tid < D_) qrow[tid] = (float)qh[row * D_ + tid];
  __syncthreads();
  if (tid < NREL) {
    float s = 0.f;
#pragma unroll
    for (int d = 0; d < D_; ++d) s += qrow[d] * relk[tid * D_ + d];
    qrel[row * NREL + tid] = s;
  }
}

// ---------------------------------------------------------------------------
// Fused attention.  One block = (b*H+h, 16-row q block).
//  - qrel strip staged by the Tensor Data Mover (TENSORcnt)
//  - k tiles double-buffered through LDS with async loads (ASYNCcnt)
//  - fp32 score strip + bf16 exp strip in LDS (~213KB of the WGP's 320KB)
__global__ void __launch_bounds__(128)
attn_kernel(const bf16* __restrict__ qh, const bf16* __restrict__ kh,
            const bf16* __restrict__ vt, const float* __restrict__ qrel,
            float* __restrict__ att_out,  // [B*H, T, T] fp32
            bf16* __restrict__ yh)        // [B*T, C] bf16 (head-merged)
{
  __shared__ float sc[16 * T_];              // 128 KB fp32 scores / exp
  __shared__ bf16 pe[16 * PES];              // 64.25 KB bf16 exp strip
  __shared__ bf16 kstage[4][2][16 * D_];     // 16 KB k staging
  __shared__ float qrel_lds[16 * NREL];      // 4.2 KB rel strip
  __shared__ float red[16 * 8];
  __shared__ float rowsum[16];
  __shared__ float rowmax[16];

  const int blk = blockIdx.x;
  const int qt = blk & 127;
  const int bh = blk >> 7;
  const int q0 = qt * 16;
  const int lane = threadIdx.x & 31;
  const int wave = threadIdx.x >> 5;

  // --- TDM: stage the contiguous 16x65 fp32 qrel strip into LDS -----------
  if (wave == 0) {
    tdm_load_1d_dwords((uint32_t)(uintptr_t)&qrel_lds[0],
                       qrel + ((size_t)bh * T_ + q0) * NREL, 16 * NREL);
    wait_tensor0();
  }

  // ---- Phase 1: scores = (q.k^T + rel gather) * 1/sqrt(D), causal mask ----
  const bf16* qbase = qh + ((size_t)bh * T_ + q0) * D_;
  v16bf aq0 = load_a_row_major(qbase + 0, D_, lane);
  v16bf aq1 = load_a_row_major(qbase + 32, D_, lane);

  const uint32_t stage_base =
      (uint32_t)(uintptr_t)&kstage[wave][0][0] + (uint32_t)(lane * 16);
  auto issue_ktile = [&](int kt, int buf) {
    const char* g = (const char*)(kh + ((size_t)bh * T_ + kt * 16) * D_) +
                    lane * 16;
    const uint32_t l = stage_base + (uint32_t)(buf * (16 * D_ * 2));
#pragma unroll
    for (int i = 0; i < 4; ++i)
      async_load_b128(l + i * 512, g + i * 512);
  };

  int buf = 0;
  if (wave <= qt) issue_ktile(wave, 0);
  __syncthreads();                           // qrel_lds visible to all waves

  for (int kt = wave; kt <= qt; kt += 4) {
    const int ktn = kt + 4;
    if (ktn <= qt) { issue_ktile(ktn, buf ^ 1); wait_async4(); }
    else           { wait_async0(); }

    v8f c = {};
    const bf16* kt_lds = &kstage[wave][buf][0];
    c = wmma_bf16(aq0, load_b_nmajor_lds(kt_lds + 0, D_, lane), c);
    c = wmma_bf16(aq1, load_b_nmajor_lds(kt_lds + 32, D_, lane), c);
    buf ^= 1;

    const int kg = kt * 16 + (lane & 15);
#pragma unroll
    for (int r = 0; r < 8; ++r) {
      const int m = (lane >> 4) * 8 + r;
      const int qg = q0 + m;
      const int idx = min(max(kg - qg, -32), 32) + 32;
      float s = (c[r] + qrel_lds[m * NREL + idx]) * 0.125f;
      if (kg > qg) s = -INFINITY;
      sc[m * T_ + kg] = s;
    }
  }
  for (int kt = qt + 1 + wave; kt < T_ / 16; kt += 4) {
    const int kg = kt * 16 + (lane & 15);
#pragma unroll
    for (int r = 0; r < 8; ++r) sc[((lane >> 4) * 8 + r) * T_ + kg] = -INFINITY;
  }
  __syncthreads();

  // ---- Phase 2: softmax stats; bf16 exp strip written once ---------------
  {
    const int row = threadIdx.x >> 3;
    const int sub = threadIdx.x & 7;
    float mx = -INFINITY;
    for (int kc = sub; kc < T_; kc += 8) mx = fmaxf(mx, sc[row * T_ + kc]);
    red[row * 8 + sub] = mx;
    __syncthreads();
    if (sub == 0) {
      float m2 = red[row * 8];
#pragma unroll
      for (int i = 1; i < 8; ++i) m2 = fmaxf(m2, red[row * 8 + i]);
      rowmax[row] = m2;
    }
    __syncthreads();
    const float rm = rowmax[row];
    float sm = 0.f;
    for (int kc = sub; kc < T_; kc += 8) {
      const float e = __expf(sc[row * T_ + kc] - rm);
      sc[row * T_ + kc] = e;
      pe[row * PES + kc] = (bf16)e;     // A-operand for phase 3, stored once
      sm += e;
    }
    red[row * 8 + sub] = sm;
    __syncthreads();
    if (sub == 0) {
      float s2 = 0.f;
#pragma unroll
      for (int i = 0; i < 8; ++i) s2 += red[row * 8 + i];
      rowsum[row] = s2;
    }
    __syncthreads();
  }
  // coalesced float4 non-temporal stream of the normalized attention strip
  for (int idx = threadIdx.x; idx < 16 * (T_ / 4); idx += 128) {
    const int row = idx >> 9;
    const int c4 = idx & 511;
    const float inv = 1.0f / rowsum[row];
    const float* s = &sc[row * T_ + c4 * 4];
    f32x4 v4 = {s[0] * inv, s[1] * inv, s[2] * inv, s[3] * inv};
    f32x4* dst = (f32x4*)(att_out + ((size_t)bh * T_ + (q0 + row)) * T_) + c4;
    __builtin_nontemporal_store(v4, dst);
  }
  __syncthreads();

  // ---- Phase 3: y = (exp @ v) / rowsum; 2x-unrolled pipeline, A from LDS --
  {
    const int nt = wave;
    const bf16* vbase = vt + ((size_t)bh * D_ + nt * 16) * T_;
    v16bf vb0 = load_b_nmajor(vbase, T_, lane);
    v16bf vb1;
    v8f c = {};
    for (int k0 = 0; k0 < T_; k0 += 64) {
      vb1 = load_b_nmajor(vbase + k0 + 32, T_, lane);
      c = wmma_bf16(load_a_pe(pe, k0, lane), vb0, c);
      if (k0 + 64 < T_) vb0 = load_b_nmajor(vbase + k0 + 64, T_, lane);
      c = wmma_bf16(load_a_pe(pe, k0 + 32, lane), vb1, c);
    }
    const int d = nt * 16 + (lane & 15);
    const int hh = bh & (H_ - 1), bb = bh >> 4;
#pragma unroll
    for (int r = 0; r < 8; ++r) {
      const int m = (lane >> 4) * 8 + r;
      const float y = c[r] / rowsum[m];
      const int t = q0 + m;
      yh[((size_t)bb * T_ + t) * C_ + hh * D_ + d] = (bf16)y;
    }
  }
}

// ---------------------------------------------------------------------------
// y_out = yh @ w_proj + b_proj (fp32), 2x-unrolled software pipeline.
__global__ void __launch_bounds__(128)
proj_gemm_kernel(const bf16* __restrict__ yb,  // [B*T, C]
                 const bf16* __restrict__ wT,  // [C, C] (transposed)
                 const float* __restrict__ bias,
                 float* __restrict__ out)      // [B*T, C]
{
  const int lane = threadIdx.x & 31;
  const int wave = threadIdx.x >> 5;
  const int m0 = (blockIdx.y * 4 + wave) * 16;
  const int n0 = blockIdx.x * 64;
  v8f c[4] = {};
  const bf16* a_base = yb + (size_t)m0 * C_;
  const bf16* b_base[4];
#pragma unroll
  for (int j = 0; j < 4; ++j) b_base[j] = wT + (size_t)(n0 + j * 16) * C_;

  v16bf a0 = load_a_row_major(a_base, C_, lane);
  v16bf b0[4];
#pragma unroll
  for (int j = 0; j < 4; ++j) b0[j] = load_b_nmajor(b_base[j], C_, lane);
  v16bf a1, b1[4];

  for (int k0 = 0; k0 < C_; k0 += 64) {
    a1 = load_a_row_major(a_base + k0 + 32, C_, lane);
#pragma unroll
    for (int j = 0; j < 4; ++j)
      b1[j] = load_b_nmajor(b_base[j] + k0 + 32, C_, lane);
#pragma unroll
    for (int j = 0; j < 4; ++j) c[j] = wmma_bf16(a0, b0[j], c[j]);
    if (k0 + 64 < C_) {
      a0 = load_a_row_major(a_base + k0 + 64, C_, lane);
#pragma unroll
      for (int j = 0; j < 4; ++j)
        b0[j] = load_b_nmajor(b_base[j] + k0 + 64, C_, lane);
    }
#pragma unroll
    for (int j = 0; j < 4; ++j) c[j] = wmma_bf16(a1, b1[j], c[j]);
  }

#pragma unroll
  for (int j = 0; j < 4; ++j) {
    const int n = n0 + j * 16 + (lane & 15);
    const float bv = bias[n];
#pragma unroll
    for (int r = 0; r < 8; ++r) {
      const int m = m0 + (lane >> 4) * 8 + r;
      out[(size_t)m * C_ + n] = c[j][r] + bv;
    }
  }
}

// ---------------------------------------------------------------------------
extern "C" void kernel_launch(void* const* d_in, const int* in_sizes, int n_in,
                              void* d_out, int out_size, void* d_ws,
                              size_t ws_size, hipStream_t stream) {
  const float* x      = (const float*)d_in[0];
  // d_in[1] = causal mask (structure hardcoded in attn_kernel)
  const float* w_attn = (const float*)d_in[2];
  const float* b_attn = (const float*)d_in[3];
  const float* w_proj = (const float*)d_in[4];
  const float* b_proj = (const float*)d_in[5];
  const float* rel_k  = (const float*)d_in[6];

  float* y_out   = (float*)d_out;                               // [B,T,C]
  float* att_out = (float*)d_out + (size_t)B_ * T_ * C_;        // [B,H,T,T]

  char* ws = (char*)d_ws;
  size_t off = 0;
  auto alloc = [&](size_t bytes) -> char* {
    char* p = ws + off;
    off += (bytes + 255) & ~(size_t)255;
    return p;
  };
  bf16*  xb   = (bf16*)alloc((size_t)B_ * T_ * C_ * sizeof(bf16));
  bf16*  waT  = (bf16*)alloc((size_t)C_ * C3_ * sizeof(bf16));
  bf16*  wpT  = (bf16*)alloc((size_t)C_ * C_ * sizeof(bf16));
  bf16*  qhb  = (bf16*)alloc((size_t)B_ * H_ * T_ * D_ * sizeof(bf16));
  bf16*  khb  = (bf16*)alloc((size_t)B_ * H_ * T_ * D_ * sizeof(bf16));
  bf16*  vtb  = (bf16*)alloc((size_t)B_ * H_ * T_ * D_ * sizeof(bf16));
  float* qrel = (float*)alloc((size_t)B_ * H_ * T_ * NREL * sizeof(float));
  bf16*  yhb  = (bf16*)alloc((size_t)B_ * T_ * C_ * sizeof(bf16));

  {
    int n1 = B_ * T_ * C_;
    f32_to_bf16_kernel<<<(n1 + 255) / 256, 256, 0, stream>>>(x, xb, n1);
    transpose_f32_to_bf16_kernel<<<dim3(C3_ / 32, C_ / 32), 256, 0, stream>>>(
        w_attn, waT, C_, C3_);
    transpose_f32_to_bf16_kernel<<<dim3(C_ / 32, C_ / 32), 256, 0, stream>>>(
        w_proj, wpT, C_, C_);
  }
  qkv_gemm_kernel<<<dim3(C3_ / 64, (B_ * T_) / 64), 128, 0, stream>>>(
      xb, waT, b_attn, qhb, khb, vtb);
  qrel_kernel<<<B_ * H_ * T_, 128, 0, stream>>>(qhb, rel_k, qrel);
  attn_kernel<<<B_ * H_ * (T_ / 16), 128, 0, stream>>>(qhb, khb, vtb, qrel,
                                                       att_out, yhb);
  proj_gemm_kernel<<<dim3(C_ / 64, (B_ * T_) / 64), 128, 0, stream>>>(
      yhb, wpT, b_proj, y_out);
}
